// Mainnet_27582279975476
// MI455X (gfx1250) — matrix operand order repaired and checked by reading, not compiled, not run on hardware
//
#include <hip/hip_runtime.h>
#include <hip/hip_bf16.h>
#include <stdint.h>

typedef __bf16 bf16;
typedef __bf16 v16bf __attribute__((ext_vector_type(16)));
typedef __bf16 v8bf  __attribute__((ext_vector_type(8)));
typedef float  v8f   __attribute__((ext_vector_type(8)));
typedef unsigned int u32x4 __attribute__((ext_vector_type(4)));
typedef int i32x4 __attribute__((ext_vector_type(4)));
typedef int i32x8 __attribute__((ext_vector_type(8)));

#define Cc 32
#define Bb 4
#define Hh 256
#define Ww 256
#define NPIX   (Bb * Hh * Ww)          // 262144
#define NTILES (NPIX / 16)             // 16384
#define NHWC_ELTS ((size_t)NPIX * Cc)  // 8388608

__device__ __forceinline__ v8f wmma_bf16(v16bf a, v16bf b, v8f c) {
    // D = A(16x32 bf16) * B(32x16 bf16) + C(16x16 f32)
    return __builtin_amdgcn_wmma_f32_16x16x32_bf16(
        /*neg_a=*/false, a, /*neg_b=*/false, b,
        /*c_mod=*/(short)0, c, /*reuse_a=*/false, /*reuse_b=*/false);
}

// ---------------------------------------------------------------------------
// TDM: issue one tensor_load_to_lds moving n_elts contiguous bf16 from global
// to LDS byte offset lds_off. D# built per CDNA5 ISA.
// Assumes the kernel's single __shared__ block sits at LDS offset 0.
// ---------------------------------------------------------------------------
__device__ __forceinline__ void tdm_load_row(const bf16* gsrc, unsigned lds_off,
                                             unsigned n_elts) {
    unsigned long long ga = (unsigned long long)(uintptr_t)gsrc;
    u32x4 g0;
    g0[0] = 1u;                                        // count=1, user descriptor
    g0[1] = lds_off;                                   // lds_addr (bytes)
    g0[2] = (unsigned)(ga & 0xffffffffu);              // global_addr[31:0]
    g0[3] = (unsigned)((ga >> 32) & 0x01ffffffu)       // global_addr[56:32]
          | (2u << 30);                                // type=2 ("image")
    i32x8 g1;
    g1[0] = (1 << 16);                                 // data_size=1 -> 2 bytes
    g1[1] = (int)((n_elts & 0xffffu) << 16);           // tensor_dim0[15:0]
    g1[2] = (int)((n_elts >> 16) & 0xffffu)            // tensor_dim0[31:16]
          | (1 << 16);                                 // tensor_dim1 = 1
    g1[3] = (int)((n_elts & 0xffffu) << 16);           // tile_dim0 = n_elts
    g1[4] = 1;                                         // tile_dim1 = 1
    g1[5] = (int)n_elts;                               // tensor_dim0_stride lo
    g1[6] = 0;
    g1[7] = 0;
    i32x4 z4 = {0, 0, 0, 0};
#if defined(__clang_major__) && (__clang_major__ >= 23)
    i32x8 z8 = {0, 0, 0, 0, 0, 0, 0, 0};
    __builtin_amdgcn_tensor_load_to_lds(g0, g1, z4, z4, z8, 0);
#else
    __builtin_amdgcn_tensor_load_to_lds(g0, g1, z4, z4, 0);
#endif
}

// ---------------------------------------------------------------------------
// K0: fold BN into conv weights, pre-swizzle all weights into WMMA B layout.
// B-matrix lane layout (32x16, 16-bit): lane L<16 -> col N=L, b[i]=K=i (i<8), 8+i (i>=8) ;
// lane L>=16 -> col N=L-16, b[i]=K=8+i (i<8), 16+i (i>=8).
// ---------------------------------------------------------------------------
#define N_WCONV (8 * 9 * 2 * 32 * 16)   // 73728
#define N_BIAS  (8 * 32)                // 256
#define N_WOUT  (18 * 32 * 16)          // 9216
#define N_WFIN  (9 * 4 * 2 * 32 * 16)   // 36864

__global__ void prep_weights(const float* __restrict__ wconv,
                             const float* __restrict__ bconv,
                             const float* __restrict__ gamma,
                             const float* __restrict__ beta,
                             const float* __restrict__ mu,
                             const float* __restrict__ var,
                             const float* __restrict__ w_outc,
                             const float* __restrict__ w_final,
                             bf16* __restrict__ wConvB,
                             float* __restrict__ biasConv,
                             bf16* __restrict__ wOutB,
                             bf16* __restrict__ wFinB) {
    const int total = N_WCONV + N_BIAS + N_WOUT + N_WFIN;
    for (int t = blockIdx.x * blockDim.x + threadIdx.x; t < total;
         t += gridDim.x * blockDim.x) {
        if (t < N_WCONV) {
            int q = t;
            int i = q & 15; q >>= 4;
            int lane = q & 31; q >>= 5;
            int cohalf = q & 1; q >>= 1;
            int tap = q % 9; q /= 9;
            int stage = q;                       // 0..7
            int co = cohalf * 16 + (lane & 15);
            int ci = i + ((i >= 8) ? 8 : 0) + ((lane < 16) ? 0 : 8);
            int kh = tap / 3, kw = tap % 3;
            int sc_idx = stage * 32 + co;        // [4][2][32] flattened
            float sc = gamma[sc_idx] * rsqrtf(var[sc_idx] + 1e-5f);
            float wv = wconv[(((stage * 32 + co) * 32 + ci) * 3 + kh) * 3 + kw];
            wConvB[t] = (bf16)(wv * sc);
        } else if (t < N_WCONV + N_BIAS) {
            int q = t - N_WCONV;
            int co = q & 31;
            int stage = q >> 5;
            int sc_idx = stage * 32 + co;
            float sc = gamma[sc_idx] * rsqrtf(var[sc_idx] + 1e-5f);
            biasConv[q] = (bconv[sc_idx] - mu[sc_idx]) * sc + beta[sc_idx];
        } else if (t < N_WCONV + N_BIAS + N_WOUT) {
            int q = t - N_WCONV - N_BIAS;
            int i = q & 15;
            int lane = (q >> 4) & 31;
            int ct = q >> 9;                     // 0..17
            int co2 = ct * 16 + (lane & 15);
            int ci = i + ((i >= 8) ? 8 : 0) + ((lane < 16) ? 0 : 8);
            wOutB[q] = (bf16)w_outc[co2 * 32 + ci];
        } else {
            int q = t - N_WCONV - N_BIAS - N_WOUT;
            int i = q & 15; int r = q >> 4;
            int lane = r & 31; r >>= 5;
            int cohalf = r & 1; r >>= 1;
            int kc = r & 3; r >>= 2;
            int tap = r;                         // 0..8
            int co = cohalf * 16 + (lane & 15);
            int ci = kc * 32 + i + ((i >= 8) ? 8 : 0) + ((lane < 16) ? 0 : 8);
            int kh = tap / 3, kw = tap % 3;
            wFinB[q] = (bf16)w_final[((co * 128 + ci) * 3 + kh) * 3 + kw];
        }
    }
}

// ---------------------------------------------------------------------------
// K1: fp32 NCHW -> bf16 NHWC
// ---------------------------------------------------------------------------
__global__ void to_nhwc_bf16(const float* __restrict__ in, bf16* __restrict__ out) {
    for (size_t idx = (size_t)blockIdx.x * blockDim.x + threadIdx.x;
         idx < NHWC_ELTS; idx += (size_t)gridDim.x * blockDim.x) {
        int c = (int)(idx & 31);
        size_t pix = idx >> 5;                    // b*H*W + y*W + x
        int x = (int)(pix % Ww);
        int y = (int)((pix / Ww) % Hh);
        int b = (int)(pix / ((size_t)Ww * Hh));
        out[idx] = (bf16)in[(((size_t)b * Cc + c) * Hh + y) * Ww + x];
    }
}

// ---------------------------------------------------------------------------
// K2: 3x3 conv 32->32, implicit GEMM via WMMA.
// Block = 256 threads (8 waves) covering a 128-px half row. TDM stages the
// 3-row x 129-px x 32-ch halo into LDS (one tensor_load_to_lds per valid
// row); waves build A fragments from LDS with ds_load_b128.
// Fused BN bias + optional residual + ReLU.
// ---------------------------------------------------------------------------
#define TILE_XS 130                       // x=-1..128 (local), stride in px
#define ROW_BYTES (TILE_XS * Cc * 2)      // 8320 B, 16B-aligned stride

__global__ void conv3x3_wmma(const bf16* __restrict__ in, bf16* __restrict__ out,
                             const bf16* __restrict__ res,
                             const bf16* __restrict__ wB,     // [9][2][32][16]
                             const float* __restrict__ bias) { // [32]
    __shared__ bf16 tile[3][TILE_XS][Cc]; // 24960 B, at LDS offset 0

    const int lane = threadIdx.x & 31;
    const int wave = threadIdx.x >> 5;
    // blockIdx.x = ((b*H + y)*2 + half)
    const int half = blockIdx.x & 1;
    const int y    = (blockIdx.x >> 1) & (Hh - 1);
    const int b    = blockIdx.x >> 9;
    const int xbase = half * 128;
    const int x0 = xbase + wave * 16;
    const int p     = lane & 15;   // A row (pixel)
    const int khalf = lane >> 4;   // K interleave half

    // ---- TDM stage: wave 0 issues up to 3 row copies, waits TENSORcnt ----
    if (wave == 0) {
        const int x_lo = (xbase == 0) ? 0 : (xbase - 1);  // clip left halo
        const int lxoff = (xbase == 0) ? 1 : 0;           // local x of x_lo
        const unsigned n_elts = 129 * Cc;                 // 129 px per row
#pragma unroll
        for (int dy = 0; dy < 3; ++dy) {
            const int yy = y + dy - 1;
            if (yy >= 0 && yy < Hh) {
                const bf16* gsrc = in + (((size_t)b * Hh + yy) * Ww + x_lo) * Cc;
                tdm_load_row(gsrc, (unsigned)(dy * ROW_BYTES + lxoff * Cc * 2),
                             n_elts);
            }
        }
        __builtin_amdgcn_s_wait_tensorcnt(0);
    }
    __syncthreads();

    // ---- WMMA main loop: A from LDS, B from pre-swizzled global ----
    v8f c0 = {}; v8f c1 = {};
#pragma unroll
    for (int tap = 0; tap < 9; ++tap) {
        const int dy = tap / 3, dx = tap % 3 - 1;
        const int yy = y + dy - 1;
        const int xx = x0 + p + dx;           // global x coordinate
        const int lx = xx - xbase + 1;        // local x in tile
        v16bf a;
        if (yy >= 0 && yy < Hh && xx >= 0 && xx < Ww) {
            v8bf lo = *(const v8bf*)(&tile[dy][lx][khalf ? 8 : 0]);
            v8bf hi = *(const v8bf*)(&tile[dy][lx][khalf ? 24 : 16]);
#pragma unroll
            for (int i = 0; i < 8; ++i) { a[i] = lo[i]; a[8 + i] = hi[i]; }
        } else {
#pragma unroll
            for (int i = 0; i < 16; ++i) a[i] = (bf16)0.0f;
        }
        const v16bf* wb = (const v16bf*)(wB + tap * 1024);
        c0 = wmma_bf16(a, wb[lane], c0);        // cout 0..15
        c1 = wmma_bf16(a, wb[32 + lane], c1);   // cout 16..31
    }

    const int coN = lane & 15;
    const float b0 = bias[coN], b1 = bias[16 + coN];
#pragma unroll
    for (int r = 0; r < 8; ++r) {
        const int prow = r + (khalf ? 8 : 0);   // C/D row layout
        const size_t addr = (((size_t)b * Hh + y) * Ww + (x0 + prow)) * Cc;
        float v0 = c0[r] + b0;
        float v1 = c1[r] + b1;
        if (res) {
            v0 += (float)res[addr + coN];
            v1 += (float)res[addr + 16 + coN];
        }
        out[addr + coN]      = (bf16)fmaxf(v0, 0.0f);
        out[addr + 16 + coN] = (bf16)fmaxf(v1, 0.0f);
    }
}

// ---------------------------------------------------------------------------
// K3: m5 = relu(m - tau); core = 1x1 conv 32->288 via 18 WMMAs into LDS;
// then 4-rate dilated kernel-prediction filter against fp32 input.
// 2 waves / block -> 36 KB LDS.
// ---------------------------------------------------------------------------
__global__ void core_kpred(const bf16* __restrict__ m,
                           const float* __restrict__ tau,
                           const bf16* __restrict__ wOutB,   // [18][32][16]
                           const float* __restrict__ b_outc, // [288]
                           const float* __restrict__ input,  // fp32 NCHW
                           bf16* __restrict__ pred) {        // NHWC, 128 ch
    __shared__ float lds[2][288][16];
    const int lane = threadIdx.x & 31;
    const int wave = threadIdx.x >> 5;
    const int tile = blockIdx.x * 2 + wave;
    if (tile >= NTILES) return;
    const int xt = tile & 15;
    const int y  = (tile >> 4) & 255;
    const int b  = tile >> 12;
    const int x0 = xt * 16;
    const int p     = lane & 15;
    const int khalf = lane >> 4;

    // Build A = m5 tile (16 px x 32 ci) with tau/ReLU applied in registers
    const bf16* base = m + (((size_t)b * Hh + y) * Ww + (x0 + p)) * Cc;
    v16bf a;
#pragma unroll
    for (int i = 0; i < 16; ++i) {
        const int ci = (i < 8) ? (i + (khalf ? 8 : 0))
                               : (16 + (i - 8) + (khalf ? 8 : 0));
        float v = (float)base[ci] - tau[ci];
        a[i] = (bf16)fmaxf(v, 0.0f);
    }

#pragma unroll
    for (int ct = 0; ct < 18; ++ct) {
        v8f c = {};
        v16bf bt = ((const v16bf*)(wOutB + ct * 512))[lane];
        c = wmma_bf16(a, bt, c);
        const int co2 = ct * 16 + (lane & 15);
        const float bo = b_outc[co2];
#pragma unroll
        for (int r = 0; r < 8; ++r)
            lds[wave][co2][r + (khalf ? 8 : 0)] = c[r] + bo;
    }
    __syncthreads();

    // Phase 2: lane = channel; adaptive 3x3 filter at rates 1..4
    const int c = lane;
    const float* inb = input + ((size_t)b * Cc + c) * Hh * Ww;
    for (int pp = 0; pp < 16; ++pp) {
        float core9[9];
#pragma unroll
        for (int t = 0; t < 9; ++t) core9[t] = lds[wave][c * 9 + t][pp];
        const int xp = x0 + pp;
#pragma unroll
        for (int rr = 0; rr < 4; ++rr) {
            const int rate = rr + 1;
            float acc = 0.0f;
#pragma unroll
            for (int t = 0; t < 9; ++t) {
                const int yy = y + (t / 3 - 1) * rate;
                const int xx = xp + (t % 3 - 1) * rate;
                const float v = (yy >= 0 && yy < Hh && xx >= 0 && xx < Ww)
                                    ? inb[yy * Ww + xx] : 0.0f;
                acc += core9[t] * v;
            }
            pred[(((size_t)b * Hh + y) * Ww + xp) * 128 + rr * 32 + c] = (bf16)acc;
        }
    }
}

// ---------------------------------------------------------------------------
// K4: final 3x3 conv 128->32 via WMMA (9 taps x 4 K-chunks x 2 co-halves),
// fp32 NCHW output.
// ---------------------------------------------------------------------------
__global__ void final_conv(const bf16* __restrict__ pred,
                           const bf16* __restrict__ wFinB,   // [9][4][2][32][16]
                           const float* __restrict__ b_final,
                           float* __restrict__ out) {
    const int lane = threadIdx.x & 31;
    const int wave = threadIdx.x >> 5;
    const int tile = blockIdx.x * (blockDim.x >> 5) + wave;
    if (tile >= NTILES) return;
    const int xt = tile & 15;
    const int y  = (tile >> 4) & 255;
    const int b  = tile >> 12;
    const int x0 = xt * 16;
    const int p     = lane & 15;
    const int khalf = lane >> 4;

    v8f c0 = {}; v8f c1 = {};
#pragma unroll
    for (int tap = 0; tap < 9; ++tap) {
        const int dy = tap / 3 - 1, dx = tap % 3 - 1;
        const int yy = y + dy, xx = x0 + p + dx;
        const bool ok = (yy >= 0 && yy < Hh && xx >= 0 && xx < Ww);
        const bf16* base = pred + (((size_t)b * Hh + yy) * Ww + xx) * 128;
#pragma unroll
        for (int kc = 0; kc < 4; ++kc) {
            v16bf a;
            if (ok) {
                v8bf lo = *(const v8bf*)(base + kc * 32 + (khalf ? 8 : 0));
                v8bf hi = *(const v8bf*)(base + kc * 32 + (khalf ? 24 : 16));
#pragma unroll
                for (int i = 0; i < 8; ++i) { a[i] = lo[i]; a[8 + i] = hi[i]; }
            } else {
#pragma unroll
                for (int i = 0; i < 16; ++i) a[i] = (bf16)0.0f;
            }
            const v16bf* wb = (const v16bf*)(wFinB + (tap * 4 + kc) * 1024);
            c0 = wmma_bf16(a, wb[lane], c0);
            c1 = wmma_bf16(a, wb[32 + lane], c1);
        }
    }

    const int coN = lane & 15;
    const float bb0 = b_final[coN], bb1 = b_final[16 + coN];
#pragma unroll
    for (int r = 0; r < 8; ++r) {
        const int prow = r + (khalf ? 8 : 0);
        out[(((size_t)b * 32 + coN) * Hh + y) * Ww + (x0 + prow)]      = c0[r] + bb0;
        out[(((size_t)b * 32 + 16 + coN) * Hh + y) * Ww + (x0 + prow)] = c1[r] + bb1;
    }
}

// ---------------------------------------------------------------------------
extern "C" void kernel_launch(void* const* d_in, const int* in_sizes, int n_in,
                              void* d_out, int out_size, void* d_ws, size_t ws_size,
                              hipStream_t stream) {
    const float* input    = (const float*)d_in[0];
    const float* wconv    = (const float*)d_in[1];
    const float* bconv    = (const float*)d_in[2];
    const float* bn_gamma = (const float*)d_in[3];
    const float* bn_beta  = (const float*)d_in[4];
    const float* bn_mean  = (const float*)d_in[5];
    const float* bn_var   = (const float*)d_in[6];
    const float* tau      = (const float*)d_in[7];
    const float* w_outc   = (const float*)d_in[8];
    const float* b_outc   = (const float*)d_in[9];
    const float* w_final  = (const float*)d_in[10];
    const float* b_final  = (const float*)d_in[11];
    float* out = (float*)d_out;

    char* ws = (char*)d_ws;
    size_t off = 0;
    auto carve = [&](size_t bytes) {
        char* p = ws + off;
        off = (off + bytes + 255) & ~(size_t)255;
        return p;
    };
    bf16*  x_nhwc   = (bf16*)carve(NHWC_ELTS * sizeof(bf16));
    bf16*  m_a      = (bf16*)carve(NHWC_ELTS * sizeof(bf16));
    bf16*  m_b      = (bf16*)carve(NHWC_ELTS * sizeof(bf16));
    bf16*  pred_cat = (bf16*)carve((size_t)NPIX * 128 * sizeof(bf16));
    bf16*  wConvB   = (bf16*)carve(N_WCONV * sizeof(bf16));
    float* biasConv = (float*)carve(N_BIAS * sizeof(float));
    bf16*  wOutB    = (bf16*)carve(N_WOUT * sizeof(bf16));
    bf16*  wFinB    = (bf16*)carve(N_WFIN * sizeof(bf16));
    (void)in_sizes; (void)n_in; (void)out_size; (void)ws_size;

    prep_weights<<<128, 256, 0, stream>>>(wconv, bconv, bn_gamma, bn_beta,
                                          bn_mean, bn_var, w_outc, w_final,
                                          wConvB, biasConv, wOutB, wFinB);
    to_nhwc_bf16<<<8192, 256, 0, stream>>>(input, x_nhwc);

    // 4 residual blocks, ping-pong across 3 NHWC buffers
    const int conv_blocks = Bb * Hh * 2;   // one block per 128-px half row
    bf16* cur = x_nhwc; bf16* tmp = m_a; bf16* oth = m_b;
    for (int blk = 0; blk < 4; ++blk) {
        conv3x3_wmma<<<conv_blocks, 256, 0, stream>>>(
            cur, tmp, (const bf16*)nullptr,
            wConvB + (size_t)(2 * blk) * 9216, biasConv + (2 * blk) * 32);
        conv3x3_wmma<<<conv_blocks, 256, 0, stream>>>(
            tmp, oth, cur,
            wConvB + (size_t)(2 * blk + 1) * 9216, biasConv + (2 * blk + 1) * 32);
        bf16* t = cur; cur = oth; oth = t;
    }

    core_kpred<<<NTILES / 2, 64, 0, stream>>>(cur, tau, wOutB, b_outc,
                                              input, pred_cat);
    final_conv<<<NTILES / 8, 256, 0, stream>>>(pred_cat, wFinB, b_final, out);
}
